// YOLOv1_38414187495679
// MI455X (gfx1250) — matrix operand order repaired and checked by reading, not compiled, not run on hardware
//
#include <hip/hip_runtime.h>
#include <hip/hip_bf16.h>
#include <math.h>

// ---------------------------------------------------------------------------
// CDNA5 (gfx1250) wave32 WMMA types
// ---------------------------------------------------------------------------
typedef __attribute__((ext_vector_type(16))) _Float16 v16h;
typedef __attribute__((ext_vector_type(8)))  _Float16 v8h;
typedef __attribute__((ext_vector_type(4)))  _Float16 v4h;
typedef __attribute__((ext_vector_type(8)))  float    v8f;

// Load 8 consecutive f16 with one b128 load.
__device__ __forceinline__ void load8h(const _Float16* __restrict__ p,
                                       v16h& d, int base)
{
    const v8h q = *(const v8h*)p;
    #pragma unroll
    for (int i = 0; i < 8; ++i) d[base + i] = q[i];
}

// Load 8 consecutive f32 as two b128 loads and convert to f16.
__device__ __forceinline__ void load8_f16(const float* __restrict__ p,
                                          v16h& d, int base)
{
    const float4 q0 = *(const float4*)(p + 0);
    const float4 q1 = *(const float4*)(p + 4);
    d[base + 0] = (_Float16)q0.x; d[base + 1] = (_Float16)q0.y;
    d[base + 2] = (_Float16)q0.z; d[base + 3] = (_Float16)q0.w;
    d[base + 4] = (_Float16)q1.x; d[base + 5] = (_Float16)q1.y;
    d[base + 6] = (_Float16)q1.z; d[base + 7] = (_Float16)q1.w;
}

// ---------------------------------------------------------------------------
// One-time weight transform: [Cout][Cin][KS*KS] f32 -> [Cout][KS*KS][Cin] f16.
// ---------------------------------------------------------------------------
__global__ void wtrans_kernel(const float* __restrict__ w,
                              _Float16* __restrict__ wt,
                              int Cin, int KSq, int total)
{
    int idx = blockIdx.x * blockDim.x + threadIdx.x;   // input-order index
    if (idx >= total) return;
    const int Kt  = Cin * KSq;
    const int m   = idx / Kt;
    const int rem = idx - m * Kt;
    const int cin = rem / KSq;
    const int r   = rem - cin * KSq;
    wt[(size_t)m * Kt + r * Cin + cin] = (_Float16)w[idx];
}

// ---------------------------------------------------------------------------
// Fast implicit-GEMM conv (Cin % 32 == 0, Cout % (64*WPB) == 0), f16 in/out.
// Block = WPB waves; all waves share one 16-pixel column tile, each wave owns
// 64 couts (4 WMMA M-tiles). Per 32-K chunk the 32(k) x 16(pix) B tile is
// staged cooperatively into double-buffered LDS (4 strided gathers + one
// ds_store_b64 per lane) and read back as b128 ds loads by every wave.
// Weights are pre-transformed to [Cout][tap][Cin] f16 -> pure b128 A loads.
// No divergent branches in the hot loop: pixel index is clamped for loads and
// only guarded at the store, so barriers are uniform and EXEC stays all-ones.
// ---------------------------------------------------------------------------
template <int KS, int WPB>
__global__ __launch_bounds__(WPB * 32)
void conv_wmma_fast(const _Float16* __restrict__ in,
                    const _Float16* __restrict__ wgtT,   // [Cout][KS*KS][Cin]
                    const float* __restrict__ bias,
                    _Float16* __restrict__ out,
                    int Cin, int H, int W,
                    int Cout, int S, int OH, int OW)
{
    __shared__ __align__(16) _Float16 ldsB[2][16][40];   // 32 k + pad, x2 buf

    const int tid    = threadIdx.x;
    const int lane   = tid & 31;
    const int waveId = tid >> 5;
    const int half   = lane >> 4;
    const int l16    = lane & 15;

    const int    npix = OH * OW;
    const int    Kt   = Cin * KS * KS;
    const size_t HW   = (size_t)H * W;
    const int nBase = blockIdx.x * 16;                   // pixel tile base
    const int mBase = blockIdx.y * (64 * WPB) + waveId * 64;
    const int batch = blockIdx.z;

    const _Float16* __restrict__ inN = in + (size_t)batch * Cin * HW;

    const int  op    = nBase + l16;                      // B column = pixel
    const bool pixOK = op < npix;
    const int  opc   = pixOK ? op : (npix - 1);          // clamped for loads
    const int  oh    = opc / OW;
    const int  ow    = opc % OW;
    const int  ihB   = oh * S;
    const int  iwB   = ow * S;

    const _Float16* __restrict__ w0 = wgtT + (size_t)(mBase + l16) * Kt;
    const size_t wStride = (size_t)16 * Kt;

    v8f acc[4] = {{}, {}, {}, {}};
    int pp = 0;                                          // LDS buffer parity

    #pragma unroll
    for (int r = 0; r < KS * KS; ++r) {
        const int kh = r / KS;                           // const after unroll
        const int kw = r % KS;
        // Per-lane gather base for this lane's pixel at this tap.
        const _Float16* __restrict__ bpTap =
            inN + (size_t)(ihB + kh) * W + (iwB + kw);

        for (int cb = 0; cb < Cin; cb += 32) {
            const int kkA = r * Cin + cb;                // GEMM-K chunk base
            __builtin_prefetch(w0 + kkA + 32, 0, 0);     // global_prefetch_b8

            // ---- cooperative B stage: quad kq covers k = kq*4..kq*4+3 ----
            #pragma unroll
            for (int kq = (tid >> 4); kq < 8; kq += 2 * WPB) {
                const _Float16* __restrict__ bq =
                    bpTap + (size_t)(cb + kq * 4) * HW;
                v4h q;
                q[0] = bq[0];
                q[1] = bq[HW];
                q[2] = bq[2 * HW];
                q[3] = bq[3 * HW];
                *(v4h*)&ldsB[pp][l16][kq * 4] = q;       // ds_store_b64
            }
            __syncthreads();

            // ---- B fragment from LDS: 2x8 f16 (k=8h.., 16+8h..) -> 2x ds b128 ----
            v16h b;
            load8h(&ldsB[pp][l16][half * 8],      b, 0);
            load8h(&ldsB[pp][l16][half * 8 + 16], b, 8);

            // ---- 4 A fragments: 2x b128 of native f16 per tile -----------
            const int kA0 = kkA + half * 8;
            #pragma unroll
            for (int t = 0; t < 4; ++t) {
                const _Float16* __restrict__ wr = w0 + (size_t)t * wStride;
                v16h a;
                load8h(wr + kA0,      a, 0);
                load8h(wr + kA0 + 16, a, 8);
                acc[t] = __builtin_amdgcn_wmma_f32_16x16x32_f16(
                             false, a, false, b, (short)0, acc[t], false, false);
            }
            pp ^= 1;                                     // double buffer
        }
    }

    // ---- store f16: lane l16 = pixel; tile t rows = mBase+t*16+half*8+r --
    if (pixOK) {
        #pragma unroll
        for (int t = 0; t < 4; ++t) {
            const int mT = mBase + t * 16 + half * 8;
            #pragma unroll
            for (int r = 0; r < 8; ++r) {
                const int m = mT + r;
                out[((size_t)batch * Cout + m) * npix + op] =
                    (_Float16)(acc[t][r] + bias[m]);
            }
        }
    }
}

// ---------------------------------------------------------------------------
// Generic (ragged-K) conv — layer 0 only (KS=7, S=2, Cin=3, Kt=147).
// f16 activations, original f32 weights. Template constants kill divisions.
// ---------------------------------------------------------------------------
template <int KS, int S>
__global__ __launch_bounds__(32)
void conv_wmma_generic(const _Float16* __restrict__ in,
                       const float* __restrict__ wgt,
                       const float* __restrict__ bias,
                       _Float16* __restrict__ out,
                       int Cin, int H, int W,
                       int Cout, int OH, int OW)
{
    const int lane = threadIdx.x;
    const int half = lane >> 4;
    const int l16  = lane & 15;

    const int npix  = OH * OW;
    const int Kt    = Cin * KS * KS;
    const int nBase = blockIdx.x * 16;
    const int mBase = blockIdx.y * 16;
    const int batch = blockIdx.z;

    const _Float16* __restrict__ inN = in + (size_t)batch * Cin * H * W;

    const int  op    = nBase + l16;
    const bool pixOK = op < npix;
    const int  opc   = pixOK ? op : (npix - 1);
    const int  oh    = opc / OW;
    const int  ow    = opc % OW;
    const int  ihB   = oh * S;
    const int  iwB   = ow * S;

    const int  mRow = mBase + l16;
    const bool mOK  = mRow < Cout;
    const float* __restrict__ wRow = wgt + (size_t)mRow * Kt;

    v8f acc = {};

    for (int kk = 0; kk < Kt; kk += 32) {
        v16h a;
        #pragma unroll
        for (int i = 0; i < 16; ++i) {
            const int kOff = (i < 8) ? (half * 8 + i)
                                     : (16 + half * 8 + (i - 8));
            const int k = kk + kOff;
            a[i] = (_Float16)((mOK && k < Kt) ? wRow[k] : 0.0f);
        }
        v16h b;
        #pragma unroll
        for (int i = 0; i < 16; ++i) {
            const int k = kk + ((i < 8) ? (half * 8 + i) : (16 + half * 8 + (i - 8)));
            _Float16 v = (_Float16)0.0f;
            if (k < Kt) {
                const int cin = k / (KS * KS);
                const int r   = k - cin * (KS * KS);
                const int kh  = r / KS;
                const int kw  = r - kh * KS;
                v = inN[((size_t)cin * H + (ihB + kh)) * W + (iwB + kw)];
            }
            b[i] = v;
        }
        acc = __builtin_amdgcn_wmma_f32_16x16x32_f16(
                  false, a, false, b, (short)0, acc, false, false);
    }

    if (pixOK) {
        const int mOff = half * 8;
        #pragma unroll
        for (int r = 0; r < 8; ++r) {
            const int m = mBase + mOff + r;
            if (m < Cout)
                out[((size_t)batch * Cout + m) * npix + op] =
                    (_Float16)(acc[r] + bias[m]);
        }
    }
}

// ---------------------------------------------------------------------------
// FC layer: out[16 x Nout] = x[16 x Kt] * W^T + bias. M = batch = 16.
// x is native f16 (b128 loads); W is f32 converted on the fly (b128 loads).
// Clamped column index -> branch-free hot loop.
// ---------------------------------------------------------------------------
template <typename OutT>
__global__ __launch_bounds__(32)
void fc_wmma_kernel(const _Float16* __restrict__ x,  // [16, Kt] f16
                    const float* __restrict__ w,     // [Nout, Kt] f32
                    const float* __restrict__ bias,  // [Nout]
                    OutT* __restrict__ out,          // [16, Nout]
                    int Kt, int Nout)
{
    const int lane  = threadIdx.x;
    const int half  = lane >> 4;
    const int l16   = lane & 15;
    const int nBase = blockIdx.x * 16;

    const _Float16* __restrict__ xRow = x + (size_t)l16 * Kt;  // A row = batch
    const int  n   = nBase + l16;                               // B col
    const bool nOK = n < Nout;
    const int  nc  = nOK ? n : (Nout - 1);
    const float* __restrict__ wRow = w + (size_t)nc * Kt;

    v8f acc = {};
    for (int kk = 0; kk < Kt; kk += 32) {      // Kt is a multiple of 32
        __builtin_prefetch(wRow + kk + 32, 0, 0);
        v16h a;
        {
            const int kA0 = kk + half * 8;
            load8h(xRow + kA0,      a, 0);
            load8h(xRow + kA0 + 16, a, 8);
        }
        v16h b;
        {
            const int kB = kk + half * 8;
            load8_f16(wRow + kB,      b, 0);
            load8_f16(wRow + kB + 16, b, 8);
        }
        acc = __builtin_amdgcn_wmma_f32_16x16x32_f16(
                  false, a, false, b, (short)0, acc, false, false);
    }

    if (nOK) {
        #pragma unroll
        for (int r = 0; r < 8; ++r) {
            const int bi = half * 8 + r;        // batch row
            out[(size_t)bi * Nout + n] = (OutT)(acc[r] + bias[n]);
        }
    }
}

// ---------------------------------------------------------------------------
// Input batch-norm (inference affine), f32 -> f16
// ---------------------------------------------------------------------------
__global__ void bn_kernel(const float* __restrict__ x, _Float16* __restrict__ y,
                          const float* __restrict__ g, const float* __restrict__ bt,
                          const float* __restrict__ mean, const float* __restrict__ var,
                          int C, int HW, int total)
{
    int idx = blockIdx.x * blockDim.x + threadIdx.x;
    if (idx >= total) return;
    int c = (idx / HW) % C;
    y[idx] = (_Float16)((x[idx] - mean[c]) * rsqrtf(var[c] + 1e-5f) * g[c] + bt[c]);
}

// ---------------------------------------------------------------------------
// 2x2/2 max pool, f16
// ---------------------------------------------------------------------------
__global__ void pool_kernel(const _Float16* __restrict__ in, _Float16* __restrict__ out,
                            int C, int H, int W, int OH, int OW, int total)
{
    int idx = blockIdx.x * blockDim.x + threadIdx.x;
    if (idx >= total) return;
    int ow = idx % OW;
    int t  = idx / OW;
    int oh = t % OH;  t /= OH;
    int c  = t % C;
    int b  = t / C;
    const _Float16* p = in + (((size_t)b * C + c) * H + oh * 2) * W + ow * 2;
    float v = fmaxf(fmaxf((float)p[0], (float)p[1]),
                    fmaxf((float)p[W], (float)p[W + 1]));
    out[idx] = (_Float16)v;
}

// ---------------------------------------------------------------------------
// YOLO head activations (f32 -> f32)
// ---------------------------------------------------------------------------
__global__ void head_kernel(const float* __restrict__ y, float* __restrict__ out, int B)
{
    const int CELL = 15, CELLS2 = 49, STRIDE = CELLS2 * CELL; // 735
    int idx = blockIdx.x * blockDim.x + threadIdx.x;
    if (idx >= B * CELLS2) return;
    int b = idx / CELLS2, cell = idx % CELLS2;
    const float* src = y   + (size_t)b * STRIDE + cell * CELL;
    float*       dst = out + (size_t)b * STRIDE + cell * CELL;
    if (cell == CELLS2 - 1) {
        for (int i = 0; i < CELL; ++i) dst[i] = src[i];
        return;
    }
    dst[0] = 1.0f / (1.0f + __expf(-src[0]));
    dst[1] = 1.0f / (1.0f + __expf(-src[1]));
    dst[2] = src[2];
    dst[3] = src[3];
    dst[4] = 1.0f / (1.0f + __expf(-src[4]));
    float mx = src[5];
    for (int i = 6; i < 15; ++i) mx = fmaxf(mx, src[i]);
    float e[10], s = 0.0f;
    for (int i = 0; i < 10; ++i) { e[i] = __expf(src[5 + i] - mx); s += e[i]; }
    float inv = 1.0f / s;
    for (int i = 0; i < 10; ++i) dst[5 + i] = e[i] * inv;
}

// ---------------------------------------------------------------------------
// Host-side orchestration
// ---------------------------------------------------------------------------
struct LayerCfg { int cin, cout, k, s; };
static const LayerCfg g_cfg[18] = {
    {3,64,7,2},{64,192,3,1},{192,128,1,1},{128,256,3,1},{256,256,1,1},
    {256,512,3,1},{512,256,1,1},{256,512,3,1},{512,256,1,1},{256,512,3,1},
    {512,512,1,1},{512,1024,3,1},{1024,512,1,1},{512,1024,3,1},{1024,512,1,1},
    {512,1024,3,1},{1024,1024,3,1},{1024,1024,3,2}
};
static const bool g_pool[18] = {
    true,true,false,false,false,true,false,false,false,false,
    false,true,false,false,false,false,false,false
};

extern "C" void kernel_launch(void* const* d_in, const int* in_sizes, int n_in,
                              void* d_out, int out_size, void* d_ws, size_t ws_size,
                              hipStream_t stream)
{
    // Input order (setup_inputs dict insertion order):
    //  [0] x ; [1..36] conv_i (w,b) ; [37..40] w1,b1,w2,b2 ; [41..44] bn g,b,m,v
    const float* x = (const float*)d_in[0];
    const float* convW[18];
    const float* convB[18];
    for (int i = 0; i < 18; ++i) {
        convW[i] = (const float*)d_in[1 + 2 * i];
        convB[i] = (const float*)d_in[2 + 2 * i];
    }
    const float* w1  = (const float*)d_in[37];
    const float* b1  = (const float*)d_in[38];
    const float* w2  = (const float*)d_in[39];
    const float* b2  = (const float*)d_in[40];
    const float* bng = (const float*)d_in[41];
    const float* bnb = (const float*)d_in[42];
    const float* bnm = (const float*)d_in[43];
    const float* bnv = (const float*)d_in[44];

    const int B = 16;

    // ---- Workspace layout: transformed f16 weights, then f16 ping-pong ---
    _Float16* wTbase = (_Float16*)d_ws;
    size_t wOff[18];
    size_t offH = 0;
    for (int i = 1; i < 18; ++i) {
        wOff[i] = offH;
        size_t n = (size_t)g_cfg[i].cout * g_cfg[i].cin * g_cfg[i].k * g_cfg[i].k;
        offH += (n + 7) & ~(size_t)7;       // keep 16B alignment
    }
    char*  actBase = (char*)d_ws + ((offH * sizeof(_Float16) + 255) & ~(size_t)255);
    size_t remBytes = ws_size - (size_t)(actBase - (char*)d_ws);
    size_t halfAct  = (remBytes / 2) & ~(size_t)255;
    _Float16* bufA = (_Float16*)actBase;
    _Float16* bufB = (_Float16*)(actBase + halfAct);

    // ---- Weight transforms (deterministic, every call) -------------------
    for (int i = 1; i < 18; ++i) {
        const LayerCfg c = g_cfg[i];
        const int total = c.cout * c.cin * c.k * c.k;
        wtrans_kernel<<<(total + 255) / 256, 256, 0, stream>>>(
            convW[i], wTbase + wOff[i], c.cin, c.k * c.k, total);
    }

    int H = 429, W = 429, C = 3;

    // ---- BatchNorm on the input (f32 -> f16) -----------------------------
    {
        int total = B * C * H * W;
        bn_kernel<<<(total + 255) / 256, 256, 0, stream>>>(
            x, bufA, bng, bnb, bnm, bnv, C, H * W, total);
    }
    _Float16* cur = bufA;
    _Float16* nxt = bufB;

    // ---- Conv stack -------------------------------------------------------
    for (int i = 0; i < 18; ++i) {
        const LayerCfg cfg = g_cfg[i];
        const int OH = (H - cfg.k) / cfg.s + 1;
        const int OW = (W - cfg.k) / cfg.s + 1;
        const int npix = OH * OW;
        const int nT   = (npix + 15) / 16;

        if (i == 0) {
            dim3 grid(nT, (cfg.cout + 15) / 16, B);
            conv_wmma_generic<7, 2><<<grid, 32, 0, stream>>>(
                cur, convW[i], convB[i], nxt, C, H, W, cfg.cout, OH, OW);
        } else if (cfg.k == 1) {
            if (cfg.cout == 128) {
                dim3 grid(nT, cfg.cout / 128, B);
                conv_wmma_fast<1, 2><<<grid, 64, 0, stream>>>(
                    cur, wTbase + wOff[i], convB[i], nxt,
                    C, H, W, cfg.cout, cfg.s, OH, OW);
            } else {
                dim3 grid(nT, cfg.cout / 256, B);
                conv_wmma_fast<1, 4><<<grid, 128, 0, stream>>>(
                    cur, wTbase + wOff[i], convB[i], nxt,
                    C, H, W, cfg.cout, cfg.s, OH, OW);
            }
        } else {
            if (cfg.cout == 192) {
                dim3 grid(nT, cfg.cout / 192, B);
                conv_wmma_fast<3, 3><<<grid, 96, 0, stream>>>(
                    cur, wTbase + wOff[i], convB[i], nxt,
                    C, H, W, cfg.cout, cfg.s, OH, OW);
            } else {
                dim3 grid(nT, cfg.cout / 256, B);
                conv_wmma_fast<3, 4><<<grid, 128, 0, stream>>>(
                    cur, wTbase + wOff[i], convB[i], nxt,
                    C, H, W, cfg.cout, cfg.s, OH, OW);
            }
        }
        { _Float16* t = cur; cur = nxt; nxt = t; }
        C = cfg.cout; H = OH; W = OW;

        if (g_pool[i]) {
            const int PH = H / 2, PW = W / 2;
            int total = B * C * PH * PW;
            pool_kernel<<<(total + 255) / 256, 256, 0, stream>>>(
                cur, nxt, C, H, W, PH, PW, total);
            { _Float16* t = cur; cur = nxt; nxt = t; }
            H = PH; W = PW;
        }
    }

    // cur holds [16, 1024] f16 (conv17 output is 1x1 spatial)
    // ---- FC1: 1024 -> 4096 (f16 out) --------------------------------------
    fc_wmma_kernel<_Float16><<<4096 / 16, 32, 0, stream>>>(
        cur, w1, b1, nxt, 1024, 4096);
    { _Float16* t = cur; cur = nxt; nxt = t; }

    // ---- FC2: 4096 -> 735 (f32 out for the head) --------------------------
    fc_wmma_kernel<float><<<(735 + 15) / 16, 32, 0, stream>>>(
        cur, w2, b2, (float*)nxt, 4096, 735);
    { _Float16* t = cur; cur = nxt; nxt = t; }

    // ---- Head activations -> d_out ----------------------------------------
    head_kernel<<<(B * 49 + 255) / 256, 256, 0, stream>>>(
        (const float*)cur, (float*)d_out, B);
}
